// FGC_23957327577330
// MI455X (gfx1250) — matrix operand order repaired and checked by reading, not compiled, not run on hardware
//
#include <hip/hip_runtime.h>
#include <math.h>
#include <stdint.h>

// ---------------------------------------------------------------------------
// CDNA5 / gfx1250: wave32, WMMA bf16 16x16x32 f32-accum, async global->LDS
// staging for conv input tiles.
// ---------------------------------------------------------------------------

typedef __attribute__((ext_vector_type(16))) __bf16 v16bf;
typedef __attribute__((ext_vector_type(8)))  float  v8f;
typedef __attribute__((ext_vector_type(8)))  unsigned short ushort8;
typedef __attribute__((ext_vector_type(4)))  int    v4i;

__device__ __forceinline__ unsigned short f2bf(float f) {
    union { float f; unsigned int u; } c; c.f = f;
    unsigned int u = c.u;
    unsigned int r = (u + 0x7FFFu + ((u >> 16) & 1u)) >> 16;  // RNE
    return (unsigned short)r;
}
__device__ __forceinline__ float bf2f(unsigned short h) {
    union { unsigned int u; float f; } c; c.u = ((unsigned int)h) << 16;
    return c.f;
}
__device__ __forceinline__ float gelu_exact(float x) {
    return 0.5f * x * (1.0f + erff(x * 0.70710678118654752f));
}
__device__ __forceinline__ float sigmoidf(float x) { return 1.0f / (1.0f + expf(-x)); }

union Frag {
    v16bf v;
    ushort8 h[2];
    unsigned int u[8];
};

// 16-byte global -> LDS copy.  Prefer the gfx1250 async DMA path
// (ASYNCcnt-tracked, no VGPR round trip); fall back to reg-staged copy.
__device__ __forceinline__ void copy16_g2l(unsigned short* dst, const unsigned short* src) {
#if __has_builtin(__builtin_amdgcn_global_load_async_to_lds_b128)
    typedef __attribute__((address_space(1))) v4i as1_v4i;
    typedef __attribute__((address_space(3))) v4i as3_v4i;
    __builtin_amdgcn_global_load_async_to_lds_b128(
        (as1_v4i*)(unsigned long long)(uintptr_t)src,
        (as3_v4i*)(unsigned int)(uintptr_t)dst, 0, 0);
#else
    *(ushort8*)dst = *(const ushort8*)src;
#endif
}
__device__ __forceinline__ void wait_async_copies() {
#if __has_builtin(__builtin_amdgcn_global_load_async_to_lds_b128)
#if __has_builtin(__builtin_amdgcn_s_wait_asynccnt)
    __builtin_amdgcn_s_wait_asynccnt(0);
#else
    asm volatile("s_wait_asynccnt 0x0" ::: "memory");
#endif
#endif
}

// ---------------------------------------------------------------------------
// Weight repack: f32 OIHW -> bf16 packed in the exact CDNA5 16-bit A-matrix
// lane layout.  Packed index = (((ocT*KPOS + kpos)*CT + ct)*32 + lane)*16 + e
// with  m = lane&15, k = (e&7) + 16*(e>>3) + 8*(lane>>4).
// ---------------------------------------------------------------------------
__global__ void repack_weights(const float* __restrict__ Wsrc,
                               unsigned short* __restrict__ dst,
                               int Cout, int Cin, int KS) {
    int total = Cout * KS * KS * Cin;
    int idx = blockIdx.x * blockDim.x + threadIdx.x;
    if (idx >= total) return;
    int e    = idx & 15;
    int lane = (idx >> 4) & 31;
    int blk  = idx >> 9;
    int CT = Cin / 32, KPOS = KS * KS;
    int ct   = blk % CT;
    int kpos = (blk / CT) % KPOS;
    int ocT  = blk / (CT * KPOS);
    int m    = lane & 15;
    int half = lane >> 4;
    int k    = (e & 7) + 16 * (e >> 3) + 8 * half;
    int oc   = ocT * 16 + m;
    int cin  = ct * 32 + k;
    int ky = kpos / KS, kx = kpos % KS;
    float w = Wsrc[((size_t)(oc * Cin + cin) * KS + ky) * KS + kx];
    dst[idx] = f2bf(w);
}

// NCHW f32 -> NHWC bf16
__global__ void nchw_to_nhwc_bf16(const float* __restrict__ src,
                                  unsigned short* __restrict__ dst,
                                  int B, int C, int H, int W) {
    int total = B * C * H * W;
    int idx = blockIdx.x * blockDim.x + threadIdx.x;
    if (idx >= total) return;
    int w = idx % W;
    int h = (idx / W) % H;
    int c = (idx / (W * H)) % C;
    int b = idx / (W * H * C);
    dst[(((size_t)b * H + h) * W + w) * C + c] = f2bf(src[idx]);
}

// ---------------------------------------------------------------------------
// Implicit-GEMM conv via WMMA with LDS-staged input window.
// Block = 256 threads = 8 waves: 4 ocTiles x 2 pixel tiles -> 64 oc x 32 px.
// Grid  = (W/32, H, B).
// Stage: KS rows x (32+KS-1) px x CIN ch of NHWC bf16 input into LDS
//        (async DMA), zero-filling padding.  Pixel stride padded by 8
//        ushorts (16B) to spread the strided B-fragment reads across banks.
// Math:  B fragments from LDS (ds_load_b128 x2), A fragments from L2.
// Epilogue: +bias, optional BN+GELU, optional *beta; NHWC bf16 store with
// stride/offset (fused concat) and/or NCHW f32 store.
// ---------------------------------------------------------------------------
template<int CIN, int KS, int PAD>
__global__ void __launch_bounds__(256)
conv_wmma(const unsigned short* __restrict__ in, int inCS,
          const unsigned short* __restrict__ Apack,
          const float* __restrict__ bias,
          const float* __restrict__ bn_g, const float* __restrict__ bn_b,
          const float* __restrict__ bn_m, const float* __restrict__ bn_v,
          const float* __restrict__ beta,
          unsigned short* __restrict__ outBF, int outCS, int outCO,
          float* __restrict__ outNCHW,
          int H, int W, int Cout) {
    constexpr int CT   = CIN / 32;
    constexpr int KPOS = KS * KS;
    constexpr int TW   = 32 + KS - 1;   // staged tile width in pixels
    constexpr int CP   = CIN + 8;       // padded per-pixel stride (ushorts)
    __shared__ unsigned short ldsB[KS * TW * CP];

    const int tid = threadIdx.x;
    const int x0  = blockIdx.x * 32;
    const int y   = blockIdx.y;
    const int b   = blockIdx.z;

    // ---- stage input window into LDS ----
    constexpr int CHUNKS = CIN / 8;          // 16B chunks per pixel
    constexpr int TOT    = KS * TW * CHUNKS;
    for (int i = tid; i < TOT; i += 256) {
        int c8 = i % CHUNKS;
        int t  = (i / CHUNKS) % TW;
        int r  = i / (CHUNKS * TW);
        int yy = y + r - PAD;
        int xx = x0 + t - PAD;
        unsigned short* dst = &ldsB[(r * TW + t) * CP + c8 * 8];
        if (yy >= 0 && yy < H && xx >= 0 && xx < W) {
            const unsigned short* src =
                in + (((size_t)(b * H + yy) * W + xx) * inCS + c8 * 8);
            copy16_g2l(dst, src);
        } else {
            ushort8 z = {0, 0, 0, 0, 0, 0, 0, 0};
            *(ushort8*)dst = z;
        }
    }
    wait_async_copies();
    __syncthreads();

    // ---- WMMA main loop (EXEC all-ones throughout) ----
    const int lane = tid & 31;
    const int wave = tid >> 5;
    const int ocT  = wave & 3;     // Cout==64 -> 4 tiles of 16
    const int pixT = wave >> 2;    // 2 pixel tiles of 16
    const int n    = lane & 15;
    const int half = lane >> 4;

    v8f acc = {0.f, 0.f, 0.f, 0.f, 0.f, 0.f, 0.f, 0.f};
    const unsigned short* Ab = Apack + ((size_t)ocT * KPOS * CT) * 512 + lane * 16;

    for (int r = 0; r < KS; ++r) {
        for (int kx = 0; kx < KS; ++kx) {
            const unsigned short* Bl =
                &ldsB[(r * TW + pixT * 16 + n + kx) * CP + 8 * half];
            const unsigned short* Ap = Ab + (size_t)((r * KS + kx) * CT) * 512;
#pragma unroll
            for (int ct = 0; ct < CT; ++ct) {
                Frag fa, fb;
                fa.v    = *reinterpret_cast<const v16bf*>(Ap + (size_t)ct * 512);
                fb.h[0] = *reinterpret_cast<const ushort8*>(Bl + ct * 32);
                fb.h[1] = *reinterpret_cast<const ushort8*>(Bl + ct * 32 + 16);
                acc = __builtin_amdgcn_wmma_f32_16x16x32_bf16(
                        false, fa.v, false, fb.v, (short)0, acc, false, false);
            }
        }
    }

    // ---- epilogue: lane holds pixel xp, channels ocT*16 + 8*half + [0..7] ----
    const int m0 = half * 8;
    const int xp = x0 + pixT * 16 + n;
    const float betav = beta ? beta[0] : 1.0f;
    float vals[8];
#pragma unroll
    for (int v = 0; v < 8; ++v) {
        int ch = ocT * 16 + m0 + v;
        float t = acc[v] + bias[ch];
        if (bn_g) {
            float s = bn_g[ch] * rsqrtf(bn_v[ch] + 1e-5f);
            t = (t - bn_m[ch]) * s + bn_b[ch];
            t = gelu_exact(t);
        }
        t *= betav;
        vals[v] = t;
    }
    if (outBF) {
        ushort8 o;
#pragma unroll
        for (int v = 0; v < 8; ++v) o[v] = f2bf(vals[v]);
        *reinterpret_cast<ushort8*>(
            outBF + ((size_t)(b * H + y) * W + xp) * outCS + outCO + ocT * 16 + m0) = o;
    }
    if (outNCHW) {
#pragma unroll
        for (int v = 0; v < 8; ++v) {
            int ch = ocT * 16 + m0 + v;
            outNCHW[(((size_t)b * Cout + ch) * H + y) * W + xp] = vals[v];
        }
    }
}

// ---------------------------------------------------------------------------
// Bilinear x2 upsample, align_corners=True
// ---------------------------------------------------------------------------
__device__ __forceinline__ float bilin_up2(const float* __restrict__ base,
                                           int Hs, int Ws, int y, int x) {
    float sy = (float)y * ((float)(Hs - 1) / (float)(2 * Hs - 1));
    float sx = (float)x * ((float)(Ws - 1) / (float)(2 * Ws - 1));
    int y0 = (int)sy; int y1 = min(y0 + 1, Hs - 1); float wy = sy - (float)y0;
    int x0 = (int)sx; int x1 = min(x0 + 1, Ws - 1); float wx = sx - (float)x0;
    float v00 = base[y0 * Ws + x0], v10 = base[y1 * Ws + x0];
    float v01 = base[y0 * Ws + x1], v11 = base[y1 * Ws + x1];
    return (v00 * (1.f - wy) + v10 * wy) * (1.f - wx) +
           (v01 * (1.f - wy) + v11 * wy) * wx;
}

// small_half (NCHW f32, 96x96) -> small (NCHW f32, 192x192) and
// channels [0..63] of the 128-channel NHWC bf16 concat buffer.
__global__ void upsample_small(const float* __restrict__ sh,
                               float* __restrict__ smallF,
                               unsigned short* __restrict__ catBF,
                               int B, int C, int Hs, int Ws) {
    int Ho = 2 * Hs, Wo = 2 * Ws;
    int total = B * C * Ho * Wo;
    int idx = blockIdx.x * blockDim.x + threadIdx.x;
    if (idx >= total) return;
    int x = idx % Wo;
    int y = (idx / Wo) % Ho;
    int c = (idx / (Wo * Ho)) % C;
    int b = idx / (Wo * Ho * C);
    const float* base = sh + ((size_t)(b * C + c) * Hs) * Ws;
    float val = bilin_up2(base, Hs, Ws, y, x);
    smallF[idx] = val;
    catBF[(((size_t)b * Ho + y) * Wo + x) * 128 + c] = f2bf(val);
}

// diff_map = ||avg2(small)-max2(x)||_2,c + ||avg2(x)-max2(small)||_2,c  @96x96
__global__ void apmp_kernel(const float* __restrict__ smallF,
                            const float* __restrict__ xF,
                            float* __restrict__ diff,
                            int B, int C, int H, int W) {
    int hp = H / 2, wp = W / 2;
    int total = B * hp * wp;
    int idx = blockIdx.x * blockDim.x + threadIdx.x;
    if (idx >= total) return;
    int pw = idx % wp;
    int ph = (idx / wp) % hp;
    int b  = idx / (wp * hp);
    float s1 = 0.f, s2 = 0.f;
    for (int c = 0; c < C; ++c) {
        const float* ps = smallF + (((size_t)(b * C + c) * H + 2 * ph) * W + 2 * pw);
        float a0 = ps[0], a1 = ps[1], a2 = ps[W], a3 = ps[W + 1];
        float aS = 0.25f * (a0 + a1 + a2 + a3);
        float mS = fmaxf(fmaxf(a0, a1), fmaxf(a2, a3));
        const float* px = xF + (((size_t)(b * C + c) * H + 2 * ph) * W + 2 * pw);
        float b0 = px[0], b1 = px[1], b2 = px[W], b3 = px[W + 1];
        float aX = 0.25f * (b0 + b1 + b2 + b3);
        float mX = fmaxf(fmaxf(b0, b1), fmaxf(b2, b3));
        float d1 = aS - mX; s1 += d1 * d1;
        float d2 = aX - mS; s2 += d2 * d2;
    }
    diff[idx] = sqrtf(s1) + sqrtf(s2);
}

// im = sigmoid(up2(in_map)); dm = sigmoid(up2(diff_small))   [B,192,192] f32
__global__ void up_sigmoid(const float* __restrict__ inm,
                           const float* __restrict__ dif,
                           float* __restrict__ im, float* __restrict__ dm,
                           int B, int Hs, int Ws) {
    int Ho = 2 * Hs, Wo = 2 * Ws;
    int total = B * Ho * Wo;
    int idx = blockIdx.x * blockDim.x + threadIdx.x;
    if (idx >= total) return;
    int x = idx % Wo;
    int y = (idx / Wo) % Ho;
    int b = idx / (Wo * Ho);
    const float* bi = inm + (size_t)b * Hs * Ws;
    const float* bd = dif + (size_t)b * Hs * Ws;
    im[idx] = sigmoidf(bilin_up2(bi, Hs, Ws, y, x));
    dm[idx] = sigmoidf(bilin_up2(bd, Hs, Ws, y, x));
}

// b_feature = x * ((dilate3(im)-im) + (dilate3(dm)-dm) + 1) -> NHWC bf16
__global__ void bfeature_kernel(const float* __restrict__ im,
                                const float* __restrict__ dm,
                                const float* __restrict__ xF,
                                unsigned short* __restrict__ bfeat,
                                int B, int C, int H, int W) {
    int total = B * H * W;
    int idx = blockIdx.x * blockDim.x + threadIdx.x;
    if (idx >= total) return;
    int x = idx % W;
    int y = (idx / W) % H;
    int b = idx / (W * H);
    float imax = -3.0e38f, dmax = -3.0e38f;
    for (int dy = -1; dy <= 1; ++dy) {
        int yy = y + dy;
        if (yy < 0 || yy >= H) continue;
        for (int dx = -1; dx <= 1; ++dx) {
            int xx = x + dx;
            if (xx < 0 || xx >= W) continue;
            size_t o = ((size_t)b * H + yy) * W + xx;
            imax = fmaxf(imax, im[o]);
            dmax = fmaxf(dmax, dm[o]);
        }
    }
    float scale = (imax - im[idx]) + (dmax - dm[idx]) + 1.0f;
    unsigned short* dst = bfeat + (size_t)idx * C;
    for (int c = 0; c < C; ++c) {
        float v = xF[(((size_t)b * C + c) * H + y) * W + x] * scale;
        dst[c] = f2bf(v);
    }
}

// conv7x7, 64 -> 1, pad 3, input NHWC bf16, weights staged in LDS.
__global__ void conv_out7(const unsigned short* __restrict__ r3,
                          const float* __restrict__ Wout,
                          const float* __restrict__ bout,
                          float* __restrict__ out,
                          int B, int C, int H, int W) {
    __shared__ float wl[49 * 64];
    for (int i = threadIdx.x; i < 49 * 64; i += blockDim.x) {
        int c = i / 49, kp = i % 49;
        wl[kp * 64 + c] = Wout[i];  // OIHW (1,64,7,7): i = c*49 + kp
    }
    __syncthreads();
    int total = B * H * W;
    int idx = blockIdx.x * blockDim.x + threadIdx.x;
    if (idx >= total) return;
    int x = idx % W;
    int y = (idx / W) % H;
    int b = idx / (W * H);
    float acc = bout[0];
    for (int ky = 0; ky < 7; ++ky) {
        int yy = y + ky - 3;
        if (yy < 0 || yy >= H) continue;
        for (int kx = 0; kx < 7; ++kx) {
            int xx = x + kx - 3;
            if (xx < 0 || xx >= W) continue;
            const unsigned short* p = r3 + ((size_t)(b * H + yy) * W + xx) * 64;
            const float* wp = wl + (ky * 7 + kx) * 64;
            for (int c0 = 0; c0 < 64; c0 += 8) {
                ushort8 hv = *reinterpret_cast<const ushort8*>(p + c0);
#pragma unroll
                for (int j = 0; j < 8; ++j) acc += bf2f(hv[j]) * wp[c0 + j];
            }
        }
    }
    out[idx] = acc;
}

// ---------------------------------------------------------------------------
extern "C" void kernel_launch(void* const* d_in, const int* in_sizes, int n_in,
                              void* d_out, int out_size, void* d_ws, size_t ws_size,
                              hipStream_t stream) {
    const int B = 8, C1 = 64, C2 = 128, H = 192, W = 192, Hs = 96, Ws = 96;

    const float* x      = (const float*)d_in[0];
    const float* smallx = (const float*)d_in[1];
    const float* inmap  = (const float*)d_in[2];
    const float* W_up  = (const float*)d_in[3];  const float* b_up  = (const float*)d_in[4];
    const float* g_up  = (const float*)d_in[5];  const float* be_up = (const float*)d_in[6];
    const float* m_up  = (const float*)d_in[7];  const float* v_up  = (const float*)d_in[8];
    const float* W_c2  = (const float*)d_in[9];  const float* b_c2  = (const float*)d_in[10];
    const float* W_d1  = (const float*)d_in[11]; const float* b_d1  = (const float*)d_in[12];
    const float* g_d1  = (const float*)d_in[13]; const float* be_d1 = (const float*)d_in[14];
    const float* m_d1  = (const float*)d_in[15]; const float* v_d1  = (const float*)d_in[16];
    const float* W_d2  = (const float*)d_in[17]; const float* b_d2  = (const float*)d_in[18];
    const float* g_d2  = (const float*)d_in[19]; const float* be_d2 = (const float*)d_in[20];
    const float* m_d2  = (const float*)d_in[21]; const float* v_d2  = (const float*)d_in[22];
    const float* W_d3  = (const float*)d_in[23]; const float* b_d3  = (const float*)d_in[24];
    const float* g_d3  = (const float*)d_in[25]; const float* be_d3 = (const float*)d_in[26];
    const float* m_d3  = (const float*)d_in[27]; const float* v_d3  = (const float*)d_in[28];
    const float* W_out = (const float*)d_in[29]; const float* b_out = (const float*)d_in[30];
    const float* beta  = (const float*)d_in[31];

    // -------- workspace carve (256B aligned) --------
    char* p = (char*)d_ws;
    auto carve = [&](size_t bytes) -> void* {
        void* r = (void*)p;
        p += (bytes + 255) & ~(size_t)255;
        return r;
    };
    unsigned short* smallx_bf  = (unsigned short*)carve((size_t)B * C2 * Hs * Ws * 2);
    unsigned short* Aup        = (unsigned short*)carve((size_t)C1 * C2 * 49 * 2);
    unsigned short* Ac2        = (unsigned short*)carve((size_t)C1 * C1 * 9 * 2);
    unsigned short* Ad1        = (unsigned short*)carve((size_t)C1 * 2 * C1 * 9 * 2);
    unsigned short* Ad2        = (unsigned short*)carve((size_t)C1 * C1 * 9 * 2);
    unsigned short* Ad3        = (unsigned short*)carve((size_t)C1 * C1 * 9 * 2);
    float*          small_half = (float*)carve((size_t)B * C1 * Hs * Ws * 4);
    float*          smallF     = (float*)carve((size_t)B * C1 * H * W * 4);
    unsigned short* catBF      = (unsigned short*)carve((size_t)B * H * W * 2 * C1 * 2);
    float*          diffS      = (float*)carve((size_t)B * Hs * Ws * 4);
    float*          im         = (float*)carve((size_t)B * H * W * 4);
    float*          dm         = (float*)carve((size_t)B * H * W * 4);
    unsigned short* bfeat      = (unsigned short*)carve((size_t)B * H * W * C1 * 2);
    unsigned short* r3         = (unsigned short*)carve((size_t)B * H * W * C1 * 2);
    unsigned short* r1         = (unsigned short*)carve((size_t)B * H * W * C1 * 2);
    unsigned short* r2         = (unsigned short*)smallF;  // smallF dead after apmp

    float* out_r   = (float*)d_out;
    float* out_map = out_r + (size_t)B * C1 * H * W;

    // -------- weight / activation repack --------
    { int t = C1 * 49 * C2;    repack_weights<<<(t + 255) / 256, 256, 0, stream>>>(W_up, Aup, C1, C2, 7); }
    { int t = C1 * 9 * C1;     repack_weights<<<(t + 255) / 256, 256, 0, stream>>>(W_c2, Ac2, C1, C1, 3); }
    { int t = C1 * 9 * 2 * C1; repack_weights<<<(t + 255) / 256, 256, 0, stream>>>(W_d1, Ad1, C1, 2 * C1, 3); }
    { int t = C1 * 9 * C1;     repack_weights<<<(t + 255) / 256, 256, 0, stream>>>(W_d2, Ad2, C1, C1, 3); }
    { int t = C1 * 9 * C1;     repack_weights<<<(t + 255) / 256, 256, 0, stream>>>(W_d3, Ad3, C1, C1, 3); }
    { int t = B * C2 * Hs * Ws;
      nchw_to_nhwc_bf16<<<(t + 255) / 256, 256, 0, stream>>>(smallx, smallx_bf, B, C2, Hs, Ws); }

    // -------- up: conv7x7 128->64 @96 + BN + GELU --------
    conv_wmma<128, 7, 3><<<dim3(Ws / 32, Hs, B), 256, 0, stream>>>(
        smallx_bf, C2, Aup, b_up, g_up, be_up, m_up, v_up, nullptr,
        nullptr, 0, 0, small_half, Hs, Ws, C1);

    // bilinear x2 -> small (f32 NCHW) + cat channels [0..63] (NHWC bf16)
    { int t = B * C1 * H * W;
      upsample_small<<<(t + 255) / 256, 256, 0, stream>>>(small_half, smallF, catBF, B, C1, Hs, Ws); }

    // AP_MP(small,x) + AP_MP(x,small) @96x96
    { int t = B * Hs * Ws;
      apmp_kernel<<<(t + 255) / 256, 256, 0, stream>>>(smallF, x, diffS, B, C1, H, W); }

    // sigmoid(up2(in_map)), sigmoid(up2(diff_map))
    { int t = B * H * W;
      up_sigmoid<<<(t + 255) / 256, 256, 0, stream>>>(inmap, diffS, im, dm, B, Hs, Ws); }

    // b_feature = x * (dilate-im + dilate-dm + 1) -> NHWC bf16
    { int t = B * H * W;
      bfeature_kernel<<<(t + 255) / 256, 256, 0, stream>>>(im, dm, x, bfeat, B, C1, H, W); }

    // fn = beta*(conv3x3 64->64 + bias) -> cat channels [64..127]
    conv_wmma<64, 3, 1><<<dim3(W / 32, H, B), 256, 0, stream>>>(
        bfeat, C1, Ac2, b_c2, nullptr, nullptr, nullptr, nullptr, beta,
        catBF, 2 * C1, C1, nullptr, H, W, C1);

    // d1: conv3x3 128->64 + BN + GELU
    conv_wmma<128, 3, 1><<<dim3(W / 32, H, B), 256, 0, stream>>>(
        catBF, 2 * C1, Ad1, b_d1, g_d1, be_d1, m_d1, v_d1, nullptr,
        r1, C1, 0, nullptr, H, W, C1);

    // d2: conv3x3 64->64 + BN + GELU
    conv_wmma<64, 3, 1><<<dim3(W / 32, H, B), 256, 0, stream>>>(
        r1, C1, Ad2, b_d2, g_d2, be_d2, m_d2, v_d2, nullptr,
        r2, C1, 0, nullptr, H, W, C1);

    // d3: conv3x3 64->64 + BN + GELU -> r (f32 NCHW to d_out) + bf16 NHWC
    conv_wmma<64, 3, 1><<<dim3(W / 32, H, B), 256, 0, stream>>>(
        r2, C1, Ad3, b_d3, g_d3, be_d3, m_d3, v_d3, nullptr,
        r3, C1, 0, out_r, H, W, C1);

    // output_map = conv7x7 64->1 + bias
    { int t = B * H * W;
      conv_out7<<<(t + 255) / 256, 256, 0, stream>>>(r3, W_out, b_out, out_map, B, C1, H, W); }

    (void)in_sizes; (void)n_in; (void)out_size; (void)ws_size;
}
